// VerifierHead_7499012899117
// MI455X (gfx1250) — matrix operand order repaired and checked by reading, not compiled, not run on hardware
//
#include <hip/hip_runtime.h>
#include <cstdint>
#include <cstring>

typedef __attribute__((ext_vector_type(16))) __bf16 v16bf;
typedef __attribute__((ext_vector_type(8)))  __bf16 v8bf;
typedef __attribute__((ext_vector_type(8)))  float  v8f;

union V16 { v16bf v; v8bf h[2]; };

static constexpr int Bt  = 64;     // batch
static constexpr int Sq  = 256;    // sequence
static constexpr int Dm  = 2048;   // hidden
static constexpr int Em  = 1024;   // E = D/2
static constexpr int H4m = 512;    // D/4
static constexpr int NVm = 3;
static constexpr int STK = 64;
static constexpr int CC  = Dm + Em;    // 3072 combined width
static constexpr int ROWS = Sq * Bt;   // 16384 (s,b) rows

__device__ __forceinline__ uint16_t f2bf_bits(float f) {
  union { float f; uint32_t u; } c; c.f = f;
  uint32_t r = c.u + 0x7fffu + ((c.u >> 16) & 1u);   // round-to-nearest-even
  return (uint16_t)(r >> 16);
}

__device__ __forceinline__ __bf16 f2bf(float f) {
  uint16_t h = f2bf_bits(f);
  __bf16 b;
  __builtin_memcpy(&b, &h, sizeof(h));
  return b;
}

// load 16 contiguous bf16 (32B) as two 128-bit transactions
__device__ __forceinline__ v16bf load16(const __bf16* p) {
  V16 t;
  t.h[0] = *(const v8bf*)(p);
  t.h[1] = *(const v8bf*)(p + 8);
  return t.v;
}

__device__ __forceinline__ uint64_t pack4(float4 f) {
  return (uint64_t)f2bf_bits(f.x)
       | ((uint64_t)f2bf_bits(f.y) << 16)
       | ((uint64_t)f2bf_bits(f.z) << 32)
       | ((uint64_t)f2bf_bits(f.w) << 48);
}

// ---------------- weight prep: fp32 -> bf16, TRANSPOSED to [n][k] ----------------
// so a lane's WMMA B-fragment (16 consecutive K at one column) is 32 contiguous bytes.
__global__ void k_prep_w1t(const float* __restrict__ w, __bf16* __restrict__ o) {
  int i = blockIdx.x * 256 + threadIdx.x;          // i = n*Dm + k over Em*Dm
  int n = i / Dm, k = i % Dm;
  o[i] = f2bf(w[(size_t)k * Em + n]);
}

__global__ void k_prep_enct(const float* __restrict__ ew, const float* __restrict__ eb,
                            __bf16* __restrict__ o, float* __restrict__ ebs) {
  int i = blockIdx.x * 256 + threadIdx.x;          // i = n*Dm + k
  int n = i / Dm, k = i % Dm;
  const size_t N = (size_t)Dm * Em;
  const size_t src = (size_t)k * Em + n;
  o[i] = f2bf(ew[src] + ew[src + N] + ew[src + 2 * N]);   // enc_w.sum(0), transposed
  if (i < Em) ebs[i] = eb[i] + eb[i + Em] + eb[i + 2 * Em];
}

__global__ void k_prep_headt(const float* __restrict__ w, __bf16* __restrict__ o) {
  int i = blockIdx.x * 256 + threadIdx.x;          // i = ((v*H4m)+n)*CC + k
  int k = i % CC;
  int vn = i / CC;
  int n = vn % H4m, v = vn / H4m;
  o[i] = f2bf(w[((size_t)v * CC + k) * H4m + n]);
}

// ---------------- phase 1: fused ctrl + enc GEMMs over all (s,b) rows ----------------
// grid: 512 blocks (one 32-row tile), 512 threads (16 waves). Each wave owns 64 cols of
// E=1024 and 2 row-fragments, so every B fragment feeds 2 WMMAs (1 b128 load per wmma).
// h=relu(xW1+b1) -> logits -> gumbel-softmax push/pop; element=(xWenc+b)/3.
__global__ __launch_bounds__(512) void k_ctrl_enc(
    const float* __restrict__ X, const float* __restrict__ gum,
    const __bf16* __restrict__ w1t, const float* __restrict__ b1,
    const float* __restrict__ w2, const float* __restrict__ b2,
    const __bf16* __restrict__ webt, const float* __restrict__ ebs,
    float* __restrict__ element, float* __restrict__ pushv, float* __restrict__ popv)
{
  __shared__ __bf16 As[32][512 + 8];   // 33 KB, row stride 1040B (16B aligned)
  __shared__ float  lsc[32][3];

  const int tile = blockIdx.x;
  const int r0 = tile * 32;        // rows r = s*64 + b ; 32-row tile has fixed s
  const int s  = r0 >> 6;
  const int b0 = r0 & 63;          // 0 or 32
  const int tid  = threadIdx.x;    // 0..511
  const int wave = tid >> 5;       // 0..15
  const int lane = tid & 31;
  const int hl   = lane >> 4;      // lane half (CDNA5 wave32 WMMA layout)
  const int m    = lane & 15;

  if (tid < 96) lsc[tid / 3][tid % 3] = 0.0f;

  const v8f zero = {0.f, 0.f, 0.f, 0.f, 0.f, 0.f, 0.f, 0.f};
  v8f acch[2][4], acce[2][4];
  #pragma unroll
  for (int rt = 0; rt < 2; ++rt)
    #pragma unroll
    for (int nt = 0; nt < 4; ++nt) { acch[rt][nt] = zero; acce[rt][nt] = zero; }

  for (int kc = 0; kc < 4; ++kc) {   // four 512-wide K chunks of D=2048
    __syncthreads();
    // stage 32x512 slab of X into LDS as bf16: float4 loads, packed 64-bit stores
    for (int idx = tid; idx < 32 * 128; idx += 512) {
      const int rr = idx >> 7, kk4 = (idx & 127) * 4;
      const float4 f = *(const float4*)&X[((size_t)(b0 + rr) * Sq + s) * Dm + kc * 512 + kk4];
      *(uint64_t*)&As[rr][kk4] = pack4(f);
    }
    __syncthreads();

    for (int kb = 0; kb < 512; kb += 32) {
      // A fragments for the two row-halves: contiguous 8-half runs -> ds_load_b128
      V16 t0, t1;
      const __bf16* ap0 = &As[m][kb + 16 * hl];
      const __bf16* ap1 = &As[m + 16][kb + 16 * hl];
      t0.h[0] = *(const v8bf*)(ap0);  t0.h[1] = *(const v8bf*)(ap0 + 8);
      t1.h[0] = *(const v8bf*)(ap1);  t1.h[1] = *(const v8bf*)(ap1 + 8);
      const v16bf a0 = t0.v, a1 = t1.v;

      const int kg = kc * 512 + kb;
      __builtin_prefetch(&w1t[(size_t)(wave * 64 + m) * Dm + kg + 32]);
      #pragma unroll
      for (int nt = 0; nt < 4; ++nt) {
        const int col = wave * 64 + nt * 16 + m;
        // B fragment: 16 consecutive K at column col -> two global_load_b128
        const v16bf bh = load16(w1t  + (size_t)col * Dm + kg + 16 * hl);
        const v16bf be = load16(webt + (size_t)col * Dm + kg + 16 * hl);
        acch[0][nt] = __builtin_amdgcn_wmma_f32_16x16x32_bf16(
            false, a0, false, bh, (short)0, acch[0][nt], false, false);
        acch[1][nt] = __builtin_amdgcn_wmma_f32_16x16x32_bf16(
            false, a1, false, bh, (short)0, acch[1][nt], false, false);
        acce[0][nt] = __builtin_amdgcn_wmma_f32_16x16x32_bf16(
            false, a0, false, be, (short)0, acce[0][nt], false, false);
        acce[1][nt] = __builtin_amdgcn_wmma_f32_16x16x32_bf16(
            false, a1, false, be, (short)0, acce[1][nt], false, false);
      }
    }
  }
  __syncthreads();

  // epilogue A: relu(h) dotted with ctrl_w2 (E x 3) -> logits partials in LDS
  float part[2][8][3];
  #pragma unroll
  for (int rt = 0; rt < 2; ++rt)
    #pragma unroll
    for (int r = 0; r < 8; ++r) { part[rt][r][0] = 0.f; part[rt][r][1] = 0.f; part[rt][r][2] = 0.f; }
  #pragma unroll
  for (int nt = 0; nt < 4; ++nt) {
    const int col = wave * 64 + nt * 16 + m;
    const float bb = b1[col];
    const float w20 = w2[col * 3 + 0], w21 = w2[col * 3 + 1], w22 = w2[col * 3 + 2];
    #pragma unroll
    for (int rt = 0; rt < 2; ++rt) {
      #pragma unroll
      for (int r = 0; r < 8; ++r) {
        const float hv = fmaxf(acch[rt][nt][r] + bb, 0.0f);
        part[rt][r][0] += hv * w20; part[rt][r][1] += hv * w21; part[rt][r][2] += hv * w22;
      }
    }
  }
  #pragma unroll
  for (int rt = 0; rt < 2; ++rt) {
    #pragma unroll
    for (int r = 0; r < 8; ++r) {
      const int row = rt * 16 + r + 8 * hl;   // C/D layout: VGPR r -> M = r + 8*laneHalf
      atomicAdd(&lsc[row][0], part[rt][r][0]);
      atomicAdd(&lsc[row][1], part[rt][r][1]);
      atomicAdd(&lsc[row][2], part[rt][r][2]);
    }
  }

  // epilogue B: element = (x·Wenc_sum + b_sum) / NV
  #pragma unroll
  for (int nt = 0; nt < 4; ++nt) {
    const int col = wave * 64 + nt * 16 + m;
    const float eb = ebs[col];
    #pragma unroll
    for (int rt = 0; rt < 2; ++rt) {
      #pragma unroll
      for (int r = 0; r < 8; ++r) {
        const int row = r0 + rt * 16 + r + 8 * hl;
        element[(size_t)row * Em + col] = (acce[rt][nt][r] + eb) * (1.0f / 3.0f);
      }
    }
  }
  __syncthreads();

  // gumbel-softmax ops (TAU=1): push/pop per row
  if (tid < 32) {
    const int b = b0 + tid;
    const float l0 = lsc[tid][0] + b2[0];
    const float l1 = lsc[tid][1] + b2[1];
    const float l2 = lsc[tid][2] + b2[2];
    const float mx = fmaxf(l0, fmaxf(l1, l2));
    const float e0 = expf(l0 - mx), e1 = expf(l1 - mx), e2 = expf(l2 - mx);
    const float iz = 1.0f / (e0 + e1 + e2);
    const float* g = gum + ((size_t)s * 64 + b) * 3;
    const float t0 = logf(e0 * iz + 1e-8f) + g[0];
    const float t1 = logf(e1 * iz + 1e-8f) + g[1];
    const float t2 = logf(e2 * iz + 1e-8f) + g[2];
    const float mx2 = fmaxf(t0, fmaxf(t1, t2));
    const float f0 = expf(t0 - mx2), f1 = expf(t1 - mx2), f2 = expf(t2 - mx2);
    const float iz2 = 1.0f / (f0 + f1 + f2);
    pushv[s * 64 + b] = f0 * iz2;
    popv [s * 64 + b] = f1 * iz2;
  }
}

// ---------------- phase 2a: sequential ptr recurrence + push-mask coefficients ----------------
__global__ void k_ptr(const float* __restrict__ pushv, const float* __restrict__ popv,
                      float* __restrict__ aw, float* __restrict__ out_ptr)
{
  const int b = threadIdx.x;
  if (b >= 64) return;
  float ptr = 0.0f;
  for (int s = 0; s < 256; ++s) {
    const float push = pushv[s * 64 + b];
    const float pop  = popv [s * 64 + b];
    const float pc = fminf(fmaxf(ptr, 0.0f), 63.0f);
    float w[64]; float sum = 0.0f;
    #pragma unroll
    for (int j = 0; j < 64; ++j) {
      const float d = (float)j - pc;
      w[j] = expf(-0.5f * d * d);
      sum += w[j];
    }
    const float sc = push / (sum + 1e-8f);
    #pragma unroll
    for (int j = 0; j < 64; ++j)
      aw[((size_t)s * 64 + b) * 64 + j] = w[j] * sc;
    ptr = fminf(fmaxf(ptr + push - pop, 0.0f), 63.0f);
  }
  out_ptr[b] = ptr;   // final ptr (B,K,1)
}

// ---------------- phase 2b: mem recurrence, stack held in VGPRs ----------------
__global__ __launch_bounds__(1024) void k_mem(
    const float* __restrict__ element, const float* __restrict__ aw,
    float* __restrict__ memsum, float* __restrict__ out_mem)
{
  const int b = blockIdx.x;
  const int e = threadIdx.x;
  __shared__ float a_sh[64];
  float mreg[64];
  #pragma unroll
  for (int j = 0; j < 64; ++j) mreg[j] = 0.0f;

  for (int s = 0; s < 256; ++s) {
    if (e < 64) a_sh[e] = aw[((size_t)s * 64 + b) * 64 + e];
    __syncthreads();
    const float el = element[((size_t)s * 64 + b) * Em + e];
    float sum = 0.0f;
    #pragma unroll
    for (int j = 0; j < 64; ++j) {
      const float a = a_sh[j];
      mreg[j] += a * (el - mreg[j]);
      sum += mreg[j];
    }
    memsum[((size_t)s * 64 + b) * Em + e] = sum * (1.0f / 64.0f);
    __syncthreads();
  }
  #pragma unroll
  for (int j = 0; j < 64; ++j)
    out_mem[((size_t)b * 64 + j) * Em + e] = mreg[j];   // final mem (B,1,64,E)
}

// ---------------- phase 3: head GEMM on [x | mem_summary], 3 heads ----------------
// 32-row tiles, 512 threads (16 waves); wave owns 32 cols of H4=512, 2 row-fragments:
// every B fragment feeds 2 WMMAs.
__global__ __launch_bounds__(512) void k_head(
    const float* __restrict__ X, const float* __restrict__ memsum,
    const __bf16* __restrict__ hw1t, const float* __restrict__ hb1,
    const float* __restrict__ hw2, const float* __restrict__ hb2,
    float* __restrict__ out_sig)
{
  __shared__ __bf16 Cs[32][512 + 8];
  __shared__ float  ssc[32][3];

  const int tile = blockIdx.x;
  const int r0 = tile * 32;
  const int s  = r0 >> 6;
  const int b0 = r0 & 63;
  const int tid  = threadIdx.x;
  const int wave = tid >> 5;
  const int lane = tid & 31;
  const int hl   = lane >> 4;
  const int m    = lane & 15;

  if (tid < 96) ssc[tid / 3][tid % 3] = 0.0f;

  const v8f zero = {0.f, 0.f, 0.f, 0.f, 0.f, 0.f, 0.f, 0.f};
  v8f acc[3][2][2];   // [head v][n-tile][row-frag]
  #pragma unroll
  for (int v = 0; v < 3; ++v)
    #pragma unroll
    for (int nt = 0; nt < 2; ++nt)
      #pragma unroll
      for (int rt = 0; rt < 2; ++rt) acc[v][nt][rt] = zero;

  for (int chunk = 0; chunk < 6; ++chunk) {   // K = [X(4x512) | memsum(2x512)]
    __syncthreads();
    if (chunk < 4) {
      for (int idx = tid; idx < 32 * 128; idx += 512) {
        const int rr = idx >> 7, kk4 = (idx & 127) * 4;
        const float4 f = *(const float4*)&X[((size_t)(b0 + rr) * Sq + s) * Dm + chunk * 512 + kk4];
        *(uint64_t*)&Cs[rr][kk4] = pack4(f);
      }
    } else {
      for (int idx = tid; idx < 32 * 128; idx += 512) {
        const int rr = idx >> 7, kk4 = (idx & 127) * 4;
        const float4 f = *(const float4*)&memsum[(size_t)(r0 + rr) * Em + (chunk - 4) * 512 + kk4];
        *(uint64_t*)&Cs[rr][kk4] = pack4(f);
      }
    }
    __syncthreads();

    for (int kb = 0; kb < 512; kb += 32) {
      V16 t0, t1;
      const __bf16* ap0 = &Cs[m][kb + 16 * hl];
      const __bf16* ap1 = &Cs[m + 16][kb + 16 * hl];
      t0.h[0] = *(const v8bf*)(ap0);  t0.h[1] = *(const v8bf*)(ap0 + 8);
      t1.h[0] = *(const v8bf*)(ap1);  t1.h[1] = *(const v8bf*)(ap1 + 8);
      const v16bf a0 = t0.v, a1 = t1.v;

      const int kg = chunk * 512 + kb;
      #pragma unroll
      for (int v = 0; v < 3; ++v) {
        #pragma unroll
        for (int nt = 0; nt < 2; ++nt) {
          const int col = wave * 32 + nt * 16 + m;
          const v16bf bf = load16(hw1t + ((size_t)v * H4m + col) * CC + kg + 16 * hl);
          acc[v][nt][0] = __builtin_amdgcn_wmma_f32_16x16x32_bf16(
              false, a0, false, bf, (short)0, acc[v][nt][0], false, false);
          acc[v][nt][1] = __builtin_amdgcn_wmma_f32_16x16x32_bf16(
              false, a1, false, bf, (short)0, acc[v][nt][1], false, false);
        }
      }
    }
  }
  __syncthreads();

  // relu(hh) · head_w2 -> scalar logit per (row, v)
  float part[2][8][3];
  #pragma unroll
  for (int rt = 0; rt < 2; ++rt)
    #pragma unroll
    for (int r = 0; r < 8; ++r) { part[rt][r][0] = 0.f; part[rt][r][1] = 0.f; part[rt][r][2] = 0.f; }
  #pragma unroll
  for (int v = 0; v < 3; ++v) {
    #pragma unroll
    for (int nt = 0; nt < 2; ++nt) {
      const int col = wave * 32 + nt * 16 + m;
      const float bb = hb1[v * H4m + col];
      const float wv = hw2[v * H4m + col];
      #pragma unroll
      for (int rt = 0; rt < 2; ++rt) {
        #pragma unroll
        for (int r = 0; r < 8; ++r) {
          const float hh = fmaxf(acc[v][nt][rt][r] + bb, 0.0f);
          part[rt][r][v] += hh * wv;
        }
      }
    }
  }
  #pragma unroll
  for (int rt = 0; rt < 2; ++rt) {
    #pragma unroll
    for (int r = 0; r < 8; ++r) {
      const int row = rt * 16 + r + 8 * hl;
      atomicAdd(&ssc[row][0], part[rt][r][0]);
      atomicAdd(&ssc[row][1], part[rt][r][1]);
      atomicAdd(&ssc[row][2], part[rt][r][2]);
    }
  }
  __syncthreads();

  if (tid < 96) {
    const int row = tid / 3, v = tid % 3;
    const float z = ssc[row][v] + hb2[v];
    const float sg = 1.0f / (1.0f + expf(-z));
    const int b = b0 + row;
    out_sig[((size_t)b * Sq + s) * 3 + v] = sg;   // (B,S,NV)
  }
}

// ---------------- phase 4: penalties ----------------
// underflow/overflow/length are identically 0 (ptr clipped to [0,63], sigmoid > 0).
__global__ void k_pen(const float* __restrict__ sig, float* __restrict__ pen)
{
  __shared__ float red[64];
  const int b = threadIdx.x;  // 64 threads
  float bal = 0.0f, invv = 0.0f;
  for (int s = 0; s < 256; ++s) {
    const float s0 = sig[((size_t)b * Sq + s) * 3 + 0];
    const float s2 = sig[((size_t)b * Sq + s) * 3 + 2];

    red[b] = (s0 - 0.5f) * (s0 - 0.5f);
    __syncthreads();
    for (int o = 32; o > 0; o >>= 1) { if (b < o) red[b] += red[b + o]; __syncthreads(); }
    if (b == 0) bal += red[0] * (1.0f / 64.0f);
    __syncthreads();

    red[b] = s2;
    __syncthreads();
    for (int o = 32; o > 0; o >>= 1) { if (b < o) red[b] += red[b + o]; __syncthreads(); }
    const float mu = red[0] * (1.0f / 64.0f);
    __syncthreads();
    red[b] = (s2 - mu) * (s2 - mu);
    __syncthreads();
    for (int o = 32; o > 0; o >>= 1) { if (b < o) red[b] += red[b + o]; __syncthreads(); }
    if (b == 0) invv += red[0] * (1.0f / 63.0f);   // ddof=1
    __syncthreads();
  }
  if (b == 0) {
    pen[0] = 0.0f;   // underflow: relu(-clip(...,0,63)) == 0
    pen[1] = 0.0f;   // overflow:  relu(clip(...)-63)    == 0
    pen[2] = bal;
    pen[3] = 0.0f;   // length: relu(-sigmoid) == 0
    pen[4] = invv;
  }
}

extern "C" void kernel_launch(void* const* d_in, const int* in_sizes, int n_in,
                              void* d_out, int out_size, void* d_ws, size_t ws_size,
                              hipStream_t stream)
{
  (void)in_sizes; (void)n_in; (void)out_size; (void)ws_size;
  const float* X    = (const float*)d_in[0];   // hidden_states (B,S,D)
  const float* gum  = (const float*)d_in[1];   // gumbel (S,K,B,3)
  const float* cw1  = (const float*)d_in[2];   // ctrl_w1 (1,D,E)
  const float* cb1  = (const float*)d_in[3];   // ctrl_b1 (1,E)
  const float* cw2  = (const float*)d_in[4];   // ctrl_w2 (1,E,3)
  const float* cb2  = (const float*)d_in[5];   // ctrl_b2 (1,3)
  const float* encw = (const float*)d_in[6];   // enc_w (3,D,E)
  const float* encb = (const float*)d_in[7];   // enc_b (3,E)
  const float* hw1  = (const float*)d_in[8];   // head_w1 (3,D+E,H4)
  const float* hb1  = (const float*)d_in[9];   // head_b1 (3,H4)
  const float* hw2  = (const float*)d_in[10];  // head_w2 (3,H4,1)
  const float* hb2  = (const float*)d_in[11];  // head_b2 (3,1)

  float* out = (float*)d_out;
  float* out_sig = out;                                    // 64*256*3
  float* out_mem = out + (size_t)Bt * Sq * NVm;            // 64*1*64*1024
  float* out_ptr = out_mem + (size_t)Bt * STK * Em;        // 64
  float* out_pen = out_ptr + Bt;                           // 5

  char* ws = (char*)d_ws;
  size_t off = 0;
  auto carve = [&](size_t bytes) -> char* {
    off = (off + 255) & ~(size_t)255;
    char* p = ws + off;
    off += bytes;
    return p;
  };
  __bf16* w1t     = (__bf16*)carve((size_t)Em * Dm * 2);        // (E, D) transposed
  __bf16* webt    = (__bf16*)carve((size_t)Em * Dm * 2);        // (E, D) transposed
  __bf16* hw1t    = (__bf16*)carve((size_t)NVm * H4m * CC * 2); // (3, H4, D+E) transposed
  float*  ebs     = (float*) carve((size_t)Em * 4);
  float*  element = (float*) carve((size_t)ROWS * Em * 4);
  float*  memsum  = (float*) carve((size_t)ROWS * Em * 4);
  float*  pushv   = (float*) carve((size_t)ROWS * 4);
  float*  popv    = (float*) carve((size_t)ROWS * 4);
  float*  aw      = (float*) carve((size_t)ROWS * STK * 4);

  k_prep_w1t  <<<(Dm * Em) / 256, 256, 0, stream>>>(cw1, w1t);
  k_prep_enct <<<(Dm * Em) / 256, 256, 0, stream>>>(encw, encb, webt, ebs);
  k_prep_headt<<<(NVm * H4m * CC) / 256, 256, 0, stream>>>(hw1, hw1t);

  k_ctrl_enc<<<ROWS / 32, 512, 0, stream>>>(X, gum, w1t, cb1, cw2, cb2, webt, ebs,
                                            element, pushv, popv);
  k_ptr<<<1, 64, 0, stream>>>(pushv, popv, aw, out_ptr);
  k_mem<<<Bt, 1024, 0, stream>>>(element, aw, memsum, out_mem);
  k_head<<<ROWS / 32, 512, 0, stream>>>(X, memsum, hw1t, hb1, hw2, hb2, out_sig);
  k_pen<<<1, 64, 0, stream>>>(out_sig, out_pen);
}
